// CNN3D_KDEEP_32770600468666
// MI455X (gfx1250) — compile-verified, matrix operand order checked by reading, not run on hardware
//
#include <hip/hip_runtime.h>
#include <hip/hip_bf16.h>

typedef _Float16 f16;
typedef __attribute__((ext_vector_type(16))) _Float16 v16h;
typedef __attribute__((ext_vector_type(8)))  float    v8f;

#define WMMA_F16(a, b, c) \
  __builtin_amdgcn_wmma_f32_16x16x32_f16(false, (a), false, (b), (short)0, (c), false, false)

// ---------------------------------------------------------------------------
// Center: sub[b] = bmin + (bmax - bmin)/2 over non-padding pos1 rows
// ---------------------------------------------------------------------------
__global__ void k_center(const float* __restrict__ pos1, float* __restrict__ sub) {
  int b = blockIdx.x, lane = threadIdx.x;
  float mx0 = -INFINITY, mx1 = -INFINITY, mx2 = -INFINITY;
  float mn0 =  INFINITY, mn1 =  INFINITY, mn2 =  INFINITY;
  for (int n = lane; n < 300; n += 32) {
    const float* p = pos1 + ((size_t)b * 300 + n) * 3;
    float x = p[0], y = p[1], z = p[2];
    float pad = ((x + y + z) == 0.0f) ? 1e10f : 0.0f;
    mx0 = fmaxf(mx0, x - pad); mx1 = fmaxf(mx1, y - pad); mx2 = fmaxf(mx2, z - pad);
    mn0 = fminf(mn0, x + pad); mn1 = fminf(mn1, y + pad); mn2 = fminf(mn2, z + pad);
  }
  for (int off = 16; off; off >>= 1) {
    mx0 = fmaxf(mx0, __shfl_xor(mx0, off, 32));
    mx1 = fmaxf(mx1, __shfl_xor(mx1, off, 32));
    mx2 = fmaxf(mx2, __shfl_xor(mx2, off, 32));
    mn0 = fminf(mn0, __shfl_xor(mn0, off, 32));
    mn1 = fminf(mn1, __shfl_xor(mn1, off, 32));
    mn2 = fminf(mn2, __shfl_xor(mn2, off, 32));
  }
  if (lane == 0) {
    sub[b * 3 + 0] = mn0 + (mx0 - mn0) * 0.5f;
    sub[b * 3 + 1] = mn1 + (mx1 - mn1) * 0.5f;
    sub[b * 3 + 2] = mn2 + (mx2 - mn2) * 0.5f;
  }
}

// ---------------------------------------------------------------------------
// Pack atom features h into WMMA-B f16 layout, once per batch.
// 12 chunks: 10 for set1 (N=300), 2 for set2 (N=40); 4 feature tiles each.
// HB[(((b*12+ch)*4+ft)*32+lane)*16+e]
// ---------------------------------------------------------------------------
__global__ void k_packh(const float* __restrict__ h1, const float* __restrict__ h2,
                        f16* __restrict__ HB) {
  int idx = blockIdx.x * blockDim.x + threadIdx.x;
  if (idx >= 16 * 12 * 4 * 512) return;
  int e = idx & 15;
  int lane = (idx >> 4) & 31;
  int ft = (idx >> 9) & 3;
  int rem = idx >> 11;
  int ch = rem % 12, b = rem / 12;
  int klo = (lane & 16) ? 8 : 0;
  int k = (e < 8) ? (klo + e) : (16 + klo + (e - 8));
  int f = ft * 16 + (lane & 15);
  const float* hsrc; int N, base;
  if (ch < 10) { hsrc = h1; N = 300; base = ch * 32; }
  else         { hsrc = h2; N = 40;  base = (ch - 10) * 32; }
  int n = base + k;
  float val = (n < N && f < 54) ? hsrc[((size_t)b * N + n) * 54 + f] : 0.0f;
  HB[idx] = (f16)val;
}

// ---------------------------------------------------------------------------
// Splat via WMMA: A = occupancy c[16 vox x 32 atoms] (computed), B = packed h.
// Output in conv layout X0[b, f, D=j, H=k, W=i].
// ---------------------------------------------------------------------------
__global__ void k_splat(const float* __restrict__ pos1, const float* __restrict__ vr1,
                        const float* __restrict__ pos2, const float* __restrict__ vr2,
                        const float* __restrict__ sub, const f16* __restrict__ HB,
                        float* __restrict__ X0) {
  __shared__ float apos[32][3];
  __shared__ float avr[32];
  int b = blockIdx.y, vt = blockIdx.x, lane = threadIdx.x;
  int m = lane & 15;
  int klo = (lane & 16) ? 8 : 0;
  int v = vt * 16 + m;
  int gi = v / 576, gj = (v / 24) % 24, gk = v % 24;
  float sx = sub[b * 3 + 0], sy = sub[b * 3 + 1], sz = sub[b * 3 + 2];
  // distance uses (pos - sub + 12) - grid  ==>  pos - (sub - 12 + grid)
  float ox = sx - 12.0f + (float)gi;
  float oy = sy - 12.0f + (float)gj;
  float oz = sz - 12.0f + (float)gk;
  v8f acc[4] = {};
  const f16* hb = HB + (size_t)b * 12 * 4 * 512;
  for (int ch = 0; ch < 12; ++ch) {
    const float* pos; const float* vrad; int N, base;
    if (ch < 10) { pos = pos1; vrad = vr1; N = 300; base = ch * 32; }
    else         { pos = pos2; vrad = vr2; N = 40;  base = (ch - 10) * 32; }
    int n = base + lane;
    if (n < N) {
      const float* p = pos + ((size_t)b * N + n) * 3;
      apos[lane][0] = p[0]; apos[lane][1] = p[1]; apos[lane][2] = p[2];
      avr[lane] = vrad[(size_t)b * N + n];
    } else {
      apos[lane][0] = 1e9f; apos[lane][1] = 1e9f; apos[lane][2] = 1e9f;
      avr[lane] = 0.0f;
    }
    __syncthreads();
    v16h a;
#pragma unroll
    for (int e = 0; e < 16; ++e) {
      int k = (e < 8) ? (klo + e) : (16 + klo + (e - 8));
      float cv = 0.0f;
      if (base + k < N) {
        float dx = apos[k][0] - ox;
        float dy = apos[k][1] - oy;
        float dz = apos[k][2] - oz;
        float d2 = dx * dx + dy * dy + dz * dz;
        float r2 = avr[k] * avr[k] / d2;
        float t = r2 * r2 * r2;
        t = t * t;                         // (vr^2/d^2)^6 == (vr/d)^12
        cv = 1.0f - __expf(-t);
      }
      a[e] = (f16)cv;
    }
#pragma unroll
    for (int ft = 0; ft < 4; ++ft) {
      v16h bm = *(const v16h*)(hb + (((size_t)ch * 4 + ft) * 32 + lane) * 16);
      acc[ft] = WMMA_F16(a, bm, acc[ft]);
    }
    __syncthreads();
  }
  int rofs = (lane >= 16) ? 8 : 0;
#pragma unroll
  for (int ft = 0; ft < 4; ++ft) {
    int f = ft * 16 + m;
    if (f >= 54) continue;
#pragma unroll
    for (int r = 0; r < 8; ++r) {
      int vox = vt * 16 + r + rofs;
      int i2 = vox / 576, j2 = (vox / 24) % 24, k2 = vox % 24;
      X0[((size_t)b * 54 + f) * 13824 + j2 * 576 + k2 * 24 + i2] = acc[ft][r];
    }
  }
}

// ---------------------------------------------------------------------------
// Weight pack: f32 [Cout][K] -> f16 WMMA A-operand layout
// ---------------------------------------------------------------------------
__global__ void k_pack(const float* __restrict__ wsrc, f16* __restrict__ dst,
                       int Cout, int K, int nk, int total) {
  int idx = blockIdx.x * blockDim.x + threadIdx.x;
  if (idx >= total) return;
  int e = idx & 15;
  int lane = (idx >> 4) & 31;
  int slot = idx >> 9;
  int kc = slot % nk;
  int ct = slot / nk;
  int co = ct * 16 + (lane & 15);
  int klo = (lane & 16) ? 8 : 0;
  int k = kc * 32 + ((e < 8) ? (klo + e) : (16 + klo + (e - 8)));
  float v = (co < Cout && k < K) ? wsrc[(size_t)co * K + k] : 0.0f;
  dst[idx] = (f16)v;
}

// ---------------------------------------------------------------------------
// K-decode table, swizzled like the B gather: one entry per (kc,lane,e).
// fire (taps=27): entry = (tap<<24) | (ci*NSP + kd*S*S + kh*S + kw)
// conv1 (taps=8): entry =             ci*13824 + kd*576 + kh*24 + kw
// invalid k -> 0xFF000000 (negative; tap=255)
// ---------------------------------------------------------------------------
__global__ void k_ktab(int* __restrict__ dst, int K, int nk, int taps, int S, int NSP,
                       int total) {
  int idx = blockIdx.x * blockDim.x + threadIdx.x;
  if (idx >= total) return;
  int e = idx & 15;
  int lane = (idx >> 4) & 31;
  int kc = idx >> 9;
  int klo = (lane & 16) ? 8 : 0;
  int k = kc * 32 + ((e < 8) ? (klo + e) : (16 + klo + (e - 8)));
  int entry = (int)0xFF000000;
  if (k < K) {
    int ci = k / taps, t = k - ci * taps;
    if (taps == 27) {
      int kd = t / 9, kh = (t % 9) / 3, kw = t % 3;
      entry = (t << 24) | (ci * NSP + (kd * S + kh) * S + kw);
    } else {
      int kd = (t >> 2), kh = (t >> 1) & 1, kw = t & 1;
      entry = ci * 13824 + kd * 576 + kh * 24 + kw;
    }
  }
  dst[idx] = entry;
}

// ---------------------------------------------------------------------------
// conv1: 2x2x2 stride-2, Cin=54 -> Cout=96, 24^3 -> 12^3, bias+ReLU
// ---------------------------------------------------------------------------
__global__ void k_conv1(const float* __restrict__ in, float* __restrict__ out,
                        const f16* __restrict__ wpack, const int* __restrict__ ktab,
                        const float* __restrict__ bias) {
  int b = blockIdx.z, st = blockIdx.x, lane = threadIdx.x;
  int m = lane & 15;
  int s = st * 16 + m;                       // 108*16 == 1728 exactly
  int d = s / 144, h = (s / 12) % 12, w = s % 12;
  int sbase = 2 * ((d * 24 + h) * 24 + w);
  const float* inb = in + (size_t)b * 54 * 13824;
  v8f acc[6] = {};
  for (int kc = 0; kc < 14; ++kc) {
    const int* tp = ktab + ((size_t)kc * 32 + lane) * 16;
    v16h bm;
#pragma unroll
    for (int e = 0; e < 16; ++e) {
      int entv = tp[e];
      float val = (entv >= 0) ? inb[sbase + entv] : 0.0f;
      bm[e] = (f16)val;
    }
#pragma unroll
    for (int i = 0; i < 6; ++i) {
      v16h a = *(const v16h*)(wpack + (((size_t)i * 14 + kc) * 32 + lane) * 16);
      acc[i] = WMMA_F16(a, bm, acc[i]);
    }
  }
  int rofs = (lane >= 16) ? 8 : 0;
#pragma unroll
  for (int i = 0; i < 6; ++i)
#pragma unroll
    for (int r = 0; r < 8; ++r) {
      int co = i * 16 + r + rofs;
      float x = acc[i][r] + bias[co];
      out[((size_t)b * 96 + co) * 1728 + s] = fmaxf(x, 0.0f);
    }
}

// ---------------------------------------------------------------------------
// Generic 3x3x3 pad-1 conv (fire squeeze/expand), implicit GEMM + WMMA.
// dup!=0 duplicates output channels (fire concat of two identical branches).
// ---------------------------------------------------------------------------
__global__ void k_fconv(const float* __restrict__ in, float* __restrict__ out,
                        const f16* __restrict__ wpack, const int* __restrict__ ktab,
                        const float* __restrict__ bias,
                        int Cin, int Cout, int S, int NSP, int nk, int dup, int outC) {
  int b = blockIdx.z, st = blockIdx.x, cg = blockIdx.y, lane = threadIdx.x;
  int m = lane & 15;
  int s = st * 16 + m;
  bool sok = s < NSP;
  int SS = S * S;
  int d = s / SS, h = (s / S) % S, w = s % S;
  int ncot = (Cout + 15) >> 4;
  const float* inb = in + (size_t)b * Cin * NSP;
  // 27-bit in-bounds mask for this lane's output position
  unsigned vmask = 0;
  if (sok) {
#pragma unroll
    for (int t = 0; t < 27; ++t) {
      int kd = t / 9, kh = (t % 9) / 3, kw = t % 3;
      unsigned dd = (unsigned)(d + kd - 1), hh = (unsigned)(h + kh - 1),
               ww = (unsigned)(w + kw - 1);
      if (dd < (unsigned)S && hh < (unsigned)S && ww < (unsigned)S) vmask |= (1u << t);
    }
  }
  int sbase = ((d - 1) * S + (h - 1)) * S + (w - 1);
  v8f acc[8] = {};
  for (int kc = 0; kc < nk; ++kc) {
    if (kc + 1 < nk)  // prefetch next A chunk (global_prefetch_b8)
      __builtin_prefetch(wpack + (((size_t)(cg * 8) * nk + (kc + 1)) * 32 + lane) * 16, 0, 3);
    const int* tp = ktab + ((size_t)kc * 32 + lane) * 16;
    v16h bm;
#pragma unroll
    for (int e = 0; e < 16; ++e) {
      int entv = tp[e];
      unsigned t = ((unsigned)entv) >> 24;       // 0..26 valid, 255 pad
      float val = 0.0f;
      if ((vmask >> (t & 31u)) & 1u) val = inb[sbase + (entv & 0x00FFFFFF)];
      bm[e] = (f16)val;
    }
#pragma unroll
    for (int i = 0; i < 8; ++i) {
      int ct = cg * 8 + i;
      if (ct < ncot) {
        const v16h* ap = (const v16h*)(wpack + (((size_t)ct * nk + kc) * 32 + lane) * 16);
        acc[i] = WMMA_F16(*ap, bm, acc[i]);
      }
    }
  }
  if (!sok) return;
  int rofs = (lane >= 16) ? 8 : 0;
#pragma unroll
  for (int i = 0; i < 8; ++i) {
    int ct = cg * 8 + i;
    if (ct >= ncot) break;
#pragma unroll
    for (int r = 0; r < 8; ++r) {
      int co = ct * 16 + r + rofs;
      if (co < Cout) {
        float x = fmaxf(acc[i][r] + bias[co], 0.0f);
        size_t o = ((size_t)b * outC + co) * NSP + s;
        out[o] = x;
        if (dup) out[o + (size_t)Cout * NSP] = x;
      }
    }
  }
}

// ---------------------------------------------------------------------------
// MaxPool3d(kernel=2, stride=3, pad=1): 12^3 -> 5^3
// ---------------------------------------------------------------------------
__global__ void k_maxpool(const float* __restrict__ in, float* __restrict__ out, int C) {
  int idx = blockIdx.x * blockDim.x + threadIdx.x;
  int total = 16 * C * 125;
  if (idx >= total) return;
  int w = idx % 5, h = (idx / 5) % 5, d = (idx / 25) % 5;
  int c = (idx / 125) % C, b = idx / (125 * C);
  const float* p = in + ((size_t)b * C + c) * 1728;
  float mx = -INFINITY;
  for (int kd = 0; kd < 2; ++kd) {
    int dd = 3 * d - 1 + kd; if (dd < 0 || dd >= 12) continue;
    for (int kh = 0; kh < 2; ++kh) {
      int hh = 3 * h - 1 + kh; if (hh < 0 || hh >= 12) continue;
      for (int kw = 0; kw < 2; ++kw) {
        int ww = 3 * w - 1 + kw; if (ww < 0 || ww >= 12) continue;
        mx = fmaxf(mx, p[(dd * 12 + hh) * 12 + ww]);
      }
    }
  }
  out[idx] = mx;
}

// ---------------------------------------------------------------------------
// AvgPool3d(kernel=3, stride=2): 5^3 -> 2^3
// ---------------------------------------------------------------------------
__global__ void k_avgpool(const float* __restrict__ in, float* __restrict__ out) {
  int idx = blockIdx.x * blockDim.x + threadIdx.x;
  if (idx >= 16 * 512 * 8) return;
  int w = idx % 2, h = (idx / 2) % 2, d = (idx / 4) % 2;
  int c = (idx / 8) % 512, b = idx / 4096;
  const float* p = in + ((size_t)b * 512 + c) * 125;
  float sacc = 0.0f;
  for (int kd = 0; kd < 3; ++kd)
    for (int kh = 0; kh < 3; ++kh)
      for (int kw = 0; kw < 3; ++kw)
        sacc += p[((2 * d + kd) * 5 + (2 * h + kh)) * 5 + (2 * w + kw)];
  out[idx] = sacc * (1.0f / 27.0f);
}

// ---------------------------------------------------------------------------
// Linear head: [16,4096] @ [4096] + b -> [16,1]
// ---------------------------------------------------------------------------
__global__ void k_linear(const float* __restrict__ x, const float* __restrict__ wlin,
                         const float* __restrict__ blin, float* __restrict__ out) {
  int b = blockIdx.x, lane = threadIdx.x;
  float sacc = 0.0f;
  for (int i = lane; i < 4096; i += 32) sacc += x[(size_t)b * 4096 + i] * wlin[i];
  for (int off = 16; off; off >>= 1) sacc += __shfl_xor(sacc, off, 32);
  if (lane == 0) out[b] = sacc + blin[0];
}

// ---------------------------------------------------------------------------
extern "C" void kernel_launch(void* const* d_in, const int* in_sizes, int n_in,
                              void* d_out, int out_size, void* d_ws, size_t ws_size,
                              hipStream_t stream) {
  (void)in_sizes; (void)n_in; (void)out_size; (void)ws_size;
  const float* pos1 = (const float*)d_in[0];
  const float* pos2 = (const float*)d_in[1];
  const float* vr1  = (const float*)d_in[2];
  const float* vr2  = (const float*)d_in[3];
  const float* h1   = (const float*)d_in[4];
  const float* h2   = (const float*)d_in[5];

  float* ws = (float*)d_ws;
  size_t off = 0;
  float* SUB = ws + off; off += 64;
  float* X0  = ws + off; off += (size_t)16 * 54 * 13824;
  float* X1  = ws + off; off += (size_t)16 * 96 * 1728;
  float* SQ  = ws + off; off += (size_t)16 * 64 * 1728;
  float* BA  = ws + off; off += (size_t)16 * 256 * 1728;
  float* BB  = ws + off; off += (size_t)16 * 256 * 1728;
  float* P5  = ws + off; off += (size_t)16 * 256 * 125;
  float* F5A = ws + off; off += (size_t)16 * 512 * 128;
  float* F5B = ws + off; off += (size_t)16 * 512 * 128;
  float* AVG = ws + off; off += (size_t)16 * 512 * 8;

  struct LayerDesc { int widx, Cout, Cin, taps, S; };
  const LayerDesc Ls[15] = {
    {6, 96, 54, 8, 12},                         // conv1 (24^3 -> 12^3)
    {8, 16, 96, 27, 12},  {10, 64, 16, 27, 12},   // fire2
    {12, 16, 128, 27, 12},{14, 64, 16, 27, 12},   // fire3
    {16, 32, 128, 27, 12},{18, 128, 32, 27, 12},  // fire4
    {20, 32, 256, 27, 5}, {22, 128, 32, 27, 5},   // fire5
    {24, 48, 256, 27, 5}, {26, 192, 48, 27, 5},   // fire6
    {28, 48, 384, 27, 5}, {30, 192, 48, 27, 5},   // fire7
    {32, 64, 384, 27, 5}, {34, 256, 64, 27, 5},   // fire8
  };
  size_t poff[15], koff[15]; int pnk[15];
  size_t pacc = 0, kacc = 0;
  for (int l = 0; l < 15; ++l) {
    int K = Ls[l].Cin * Ls[l].taps;
    int nk = (K + 31) / 32;
    int ncot = (Ls[l].Cout + 15) / 16;
    poff[l] = pacc; pnk[l] = nk;
    pacc += (size_t)ncot * nk * 512;
    koff[l] = kacc;
    kacc += (size_t)nk * 512;
  }
  f16* PACK = (f16*)(ws + off);                         // packed weights (f16)
  f16* HB   = PACK + ((pacc + 63) & ~(size_t)63);       // packed splat features
  int* KTAB = (int*)(HB + (size_t)16 * 12 * 4 * 512 + 64);

  for (int l = 0; l < 15; ++l) {
    int K = Ls[l].Cin * Ls[l].taps;
    int nk = pnk[l];
    int ncot = (Ls[l].Cout + 15) / 16;
    size_t ptot = (size_t)ncot * nk * 512;
    k_pack<<<(unsigned)((ptot + 255) / 256), 256, 0, stream>>>(
        (const float*)d_in[Ls[l].widx], PACK + poff[l], Ls[l].Cout, K, nk, (int)ptot);
    int NSP = Ls[l].S * Ls[l].S * Ls[l].S;
    size_t ktot = (size_t)nk * 512;
    k_ktab<<<(unsigned)((ktot + 255) / 256), 256, 0, stream>>>(
        KTAB + koff[l], K, nk, Ls[l].taps, Ls[l].S, NSP, (int)ktot);
  }
  { int tot = 16 * 12 * 4 * 512;
    k_packh<<<(tot + 255) / 256, 256, 0, stream>>>(h1, h2, HB); }

  k_center<<<16, 32, 0, stream>>>(pos1, SUB);
  k_splat<<<dim3(864, 16), 32, 0, stream>>>(pos1, vr1, pos2, vr2, SUB, HB, X0);
  k_conv1<<<dim3(108, 1, 16), 32, 0, stream>>>(X0, X1, PACK + poff[0], KTAB + koff[0],
                                               (const float*)d_in[7]);

  auto fconv = [&](const float* in, float* out, int l, int Cin, int Cout, int S,
                   int dup, int outC, int bidx) {
    int NSP = S * S * S;
    int ncot = (Cout + 15) / 16, ngrp = (ncot + 7) / 8;
    int ntile = (NSP + 15) / 16;
    k_fconv<<<dim3(ntile, ngrp, 16), 32, 0, stream>>>(
        in, out, PACK + poff[l], KTAB + koff[l], (const float*)d_in[bidx],
        Cin, Cout, S, NSP, pnk[l], dup, outC);
  };

  fconv(X1, SQ, 1, 96, 16, 12, 0, 16, 9);      // fire2 squeeze
  fconv(SQ, BA, 2, 16, 64, 12, 1, 128, 11);    // fire2 expand (dup)
  fconv(BA, SQ, 3, 128, 16, 12, 0, 16, 13);    // fire3 squeeze
  fconv(SQ, BB, 4, 16, 64, 12, 1, 128, 15);    // fire3 expand
  fconv(BB, SQ, 5, 128, 32, 12, 0, 32, 17);    // fire4 squeeze
  fconv(SQ, BA, 6, 32, 128, 12, 1, 256, 19);   // fire4 expand

  { int tot = 16 * 256 * 125;
    k_maxpool<<<(tot + 255) / 256, 256, 0, stream>>>(BA, P5, 256); }

  fconv(P5, SQ, 7, 256, 32, 5, 0, 32, 21);     // fire5 squeeze
  fconv(SQ, F5A, 8, 32, 128, 5, 1, 256, 23);   // fire5 expand
  fconv(F5A, SQ, 9, 256, 48, 5, 0, 48, 25);    // fire6 squeeze
  fconv(SQ, F5B, 10, 48, 192, 5, 1, 384, 27);  // fire6 expand
  fconv(F5B, SQ, 11, 384, 48, 5, 0, 48, 29);   // fire7 squeeze
  fconv(SQ, F5A, 12, 48, 192, 5, 1, 384, 31);  // fire7 expand
  fconv(F5A, SQ, 13, 384, 64, 5, 0, 64, 33);   // fire8 squeeze
  fconv(SQ, F5B, 14, 64, 256, 5, 1, 512, 35);  // fire8 expand

  { int tot = 16 * 512 * 8;
    k_avgpool<<<(tot + 255) / 256, 256, 0, stream>>>(F5B, AVG); }

  k_linear<<<16, 32, 0, stream>>>(AVG, (const float*)d_in[36], (const float*)d_in[37],
                                  (float*)d_out);
}
